// SelfAttentionHead_46797963657532
// MI455X (gfx1250) — matrix operand order repaired and checked by reading, not compiled, not run on hardware
//
#include <hip/hip_runtime.h>
#include <hip/hip_bf16.h>

// ---------------------------------------------------------------------------
// Self-attention head, S=4096, E=V=1024, fp32 in/out.
//   1) cast x -> bf16 ; cast+transpose Wq,Wk,Wv -> bf16 [V,E]
//   2) Q=x@Wq, K=x@Wk (bf16 [S,V]);  V=x@Wv stored TRANSPOSED (bf16 [V,S])
//   3) scores = Q @ K^T                  (f32 [S,S])
//   4) row softmax (f32), probs written in-place as bf16 (row stride 2S)
//   5) z = P @ V  (B = V^T via BT path)  (f32 -> d_out)
// All GEMMs share one kernel: C = A[M,K] * B^T with B given [N,K] row-major.
// 128x128x32 block tile, 8 wave32 (4m x 2n), 32x64/wave = 8 WMMA per K-step,
// LDS double-buffered staging (single barrier per K-step).
// ---------------------------------------------------------------------------

#define S_LEN 4096
#define E_DIM 1024
#define V_DIM 1024

typedef __attribute__((ext_vector_type(8)))  __bf16 bf16x8;
typedef __attribute__((ext_vector_type(16))) __bf16 bf16x16;
typedef __attribute__((ext_vector_type(8)))  float  f32x8;

__device__ __forceinline__ unsigned short f32_to_bf16_rne(float f) {
    union { float f; unsigned u; } c; c.f = f;
    unsigned u = c.u;
    unsigned r = (u + 0x7FFFu + ((u >> 16) & 1u)) >> 16;
    return (unsigned short)r;
}

__global__ __launch_bounds__(256) void cast_f32_to_bf16(const float4* __restrict__ in,
                                                        ushort4* __restrict__ out,
                                                        int n4) {
    int i = blockIdx.x * blockDim.x + threadIdx.x;
    if (i < n4) {
        float4 v = in[i];
        ushort4 o;
        o.x = f32_to_bf16_rne(v.x);
        o.y = f32_to_bf16_rne(v.y);
        o.z = f32_to_bf16_rne(v.z);
        o.w = f32_to_bf16_rne(v.w);
        out[i] = o;
    }
}

// out[c][r] = bf16(in[r][c]) ; rows x cols fp32 in, [cols][rows] bf16 out.
__global__ __launch_bounds__(256) void cast_transpose_f32_bf16(
    const float* __restrict__ in, unsigned short* __restrict__ out,
    int rows, int cols) {
    __shared__ unsigned short tile[32][33];
    const int tx = threadIdx.x & 31;
    const int ty = threadIdx.x >> 5;           // 0..7
    const int r0 = blockIdx.y * 32;
    const int c0 = blockIdx.x * 32;
    #pragma unroll
    for (int i = 0; i < 4; ++i) {
        int r = r0 + ty + i * 8;
        tile[ty + i * 8][tx] = f32_to_bf16_rne(in[(size_t)r * cols + c0 + tx]);
    }
    __syncthreads();
    #pragma unroll
    for (int i = 0; i < 4; ++i) {
        int c = c0 + ty + i * 8;
        out[(size_t)c * rows + r0 + tx] = tile[tx][ty + i * 8];
    }
}

// Build a 16-element bf16 fragment from two 16B-aligned LDS chunks.
__device__ __forceinline__ bf16x16 make_frag(const __bf16* p0, const __bf16* p1) {
    bf16x8 lo = *(const bf16x8*)p0;
    bf16x8 hi = *(const bf16x8*)p1;
    return __builtin_shufflevector(lo, hi, 0,1,2,3,4,5,6,7,8,9,10,11,12,13,14,15);
}

// ---------------------------------------------------------------------------
// C[M,N] = A[M,K] * B^T,  B given as [N,K] row-major.
// OM: 0 = f32 row-major, 1 = bf16 row-major, 2 = bf16 transposed ([N,M], packed)
// ---------------------------------------------------------------------------
template <int OM>
__global__ __launch_bounds__(256) void gemm_bt_bf16_wmma(
    const unsigned short* __restrict__ A, int lda,
    const unsigned short* __restrict__ B, int ldb,
    void* __restrict__ C, int ldc,
    int M, int N, int K)
{
    constexpr int BM = 128, BN = 128, BK = 32;
    constexpr int STR = BK + 8;              // 40 elems = 80B row stride (16B aligned)

    __shared__ __bf16 As[2][BM * STR];       // [row][k]
    __shared__ __bf16 Bs[2][BN * STR];       // [n][k]

    const int tid  = threadIdx.x;
    const int lane = tid & 31;
    const int wave = tid >> 5;
    const int wm   = wave & 3;               // 0..3 : 32-row strip
    const int wn   = wave >> 2;              // 0..1 : 64-col strip
    const int row0 = blockIdx.y * BM;
    const int col0 = blockIdx.x * BN;

    // staging geometry: 512 x 16B chunks per tile, 2 per thread
    const int r_a0 = (tid) >> 2,       kc_a0 = ((tid) & 3) * 8;
    const int r_a1 = (tid + 256) >> 2, kc_a1 = ((tid + 256) & 3) * 8;

    f32x8 acc[2][4] = {};

    // ---- prologue: stage slab 0 into buffer 0
    {
        uint4 a0 = *(const uint4*)(A + (size_t)(row0 + r_a0) * lda + kc_a0);
        uint4 a1 = *(const uint4*)(A + (size_t)(row0 + r_a1) * lda + kc_a1);
        uint4 b0 = *(const uint4*)(B + (size_t)(col0 + r_a0) * ldb + kc_a0);
        uint4 b1 = *(const uint4*)(B + (size_t)(col0 + r_a1) * ldb + kc_a1);
        *(uint4*)&As[0][r_a0 * STR + kc_a0] = a0;
        *(uint4*)&As[0][r_a1 * STR + kc_a1] = a1;
        *(uint4*)&Bs[0][r_a0 * STR + kc_a0] = b0;
        *(uint4*)&Bs[0][r_a1 * STR + kc_a1] = b1;
    }
    __syncthreads();

    int p = 0;
    for (int k0 = 0; k0 < K; k0 += BK, p ^= 1) {
        const bool more = (k0 + BK) < K;

        // ---- issue next slab's global loads early (land during WMMAs)
        uint4 a0, a1, b0, b1;
        if (more) {
            int kn = k0 + BK;
            a0 = *(const uint4*)(A + (size_t)(row0 + r_a0) * lda + kn + kc_a0);
            a1 = *(const uint4*)(A + (size_t)(row0 + r_a1) * lda + kn + kc_a1);
            b0 = *(const uint4*)(B + (size_t)(col0 + r_a0) * ldb + kn + kc_a0);
            b1 = *(const uint4*)(B + (size_t)(col0 + r_a1) * ldb + kn + kc_a1);
        }
        if (k0 + 2 * BK < K) {
            __builtin_prefetch(A + (size_t)(row0 + r_a0) * lda + k0 + 2 * BK + kc_a0, 0, 1);
            __builtin_prefetch(B + (size_t)(col0 + r_a0) * ldb + k0 + 2 * BK + kc_a0, 0, 1);
        }

        // ---- fragments per ISA 16-bit VGPR layouts
        // A 16x32: lanes 0-15 hold K 0-7 & 16-23; lanes 16-31 hold K 8-15 & 24-31
        // B 32x16: lanes 0-15 hold K 0-7 & 16-23; lanes 16-31 hold K 8-15 & 24-31
        const int kbA = (lane < 16) ? 0 : 8;
        const int kbB = (lane < 16) ? 0 : 8;
        bf16x16 afrag[2], bfrag[4];
        #pragma unroll
        for (int tm = 0; tm < 2; ++tm) {
            int r = wm * 32 + tm * 16 + (lane & 15);
            afrag[tm] = make_frag(&As[p][r * STR + kbA], &As[p][r * STR + kbA + 16]);
        }
        #pragma unroll
        for (int tn = 0; tn < 4; ++tn) {
            int n = wn * 64 + tn * 16 + (lane & 15);
            bfrag[tn] = make_frag(&Bs[p][n * STR + kbB], &Bs[p][n * STR + kbB + 16]);
        }

        #pragma unroll
        for (int tm = 0; tm < 2; ++tm)
            #pragma unroll
            for (int tn = 0; tn < 4; ++tn)
                acc[tm][tn] = __builtin_amdgcn_wmma_f32_16x16x32_bf16(
                    false, afrag[tm], false, bfrag[tn],
                    (short)0, acc[tm][tn], false, false);

        // ---- stage next slab into the other buffer (no barrier crossing)
        if (more) {
            *(uint4*)&As[p ^ 1][r_a0 * STR + kc_a0] = a0;
            *(uint4*)&As[p ^ 1][r_a1 * STR + kc_a1] = a1;
            *(uint4*)&Bs[p ^ 1][r_a0 * STR + kc_a0] = b0;
            *(uint4*)&Bs[p ^ 1][r_a1 * STR + kc_a1] = b1;
        }
        __syncthreads();
    }

    // ---- store C: VGPR i -> M=i (lanes 0-15) / M=i+8 (lanes 16-31), N=lane%16
    const int mlo = (lane >> 4) * 8;
    const int nl  = lane & 15;
    #pragma unroll
    for (int tm = 0; tm < 2; ++tm) {
        #pragma unroll
        for (int tn = 0; tn < 4; ++tn) {
            const int mbase = row0 + wm * 32 + tm * 16 + mlo;
            const int n     = col0 + wn * 64 + tn * 16 + nl;
            if (OM == 2) {
                // transposed bf16: lane's 8 acc entries are 8 consecutive M at
                // fixed N -> one packed 16B store to C[n][mbase..mbase+7]
                unsigned short pk[8];
                #pragma unroll
                for (int i = 0; i < 8; ++i) pk[i] = f32_to_bf16_rne(acc[tm][tn][i]);
                *(uint4*)((unsigned short*)C + (size_t)n * ldc + mbase) = *(const uint4*)pk;
            } else {
                #pragma unroll
                for (int i = 0; i < 8; ++i) {
                    float v = acc[tm][tn][i];
                    if (OM == 1)
                        ((unsigned short*)C)[(size_t)(mbase + i) * ldc + n] = f32_to_bf16_rne(v);
                    else
                        ((float*)C)[(size_t)(mbase + i) * ldc + n] = v;
                }
            }
        }
    }
}

// ---------------------------------------------------------------------------
// Row softmax over fp32 scores [S,S]; writes bf16 probabilities IN PLACE over
// the start of each fp32 row (so P has element row stride 2*S as bf16).
// ---------------------------------------------------------------------------
__global__ __launch_bounds__(256) void softmax_rows_inplace(float* __restrict__ scores, int n) {
    const int tid = threadIdx.x;
    float* rowp = scores + (size_t)blockIdx.x * n;

    float vals[16];
    float m = -INFINITY;
    #pragma unroll
    for (int i = 0; i < 16; ++i) {
        vals[i] = rowp[tid + i * 256];
        m = fmaxf(m, vals[i]);
    }

    __shared__ float red[256];
    red[tid] = m;
    __syncthreads();
    for (int s = 128; s > 0; s >>= 1) {
        if (tid < s) red[tid] = fmaxf(red[tid], red[tid + s]);
        __syncthreads();
    }
    m = red[0];
    __syncthreads();

    float sum = 0.f;
    #pragma unroll
    for (int i = 0; i < 16; ++i) {
        vals[i] = __expf(vals[i] - m);
        sum += vals[i];
    }
    red[tid] = sum;
    __syncthreads();
    for (int s = 128; s > 0; s >>= 1) {
        if (tid < s) red[tid] += red[tid + s];
        __syncthreads();
    }
    const float inv = 1.0f / red[0];
    __syncthreads();  // all reads of rowp done before aliasing bf16 writes

    unsigned short* outp = (unsigned short*)rowp;   // bf16 view, in place
    #pragma unroll
    for (int i = 0; i < 16; ++i)
        outp[tid + i * 256] = f32_to_bf16_rne(vals[i] * inv);
}

// ---------------------------------------------------------------------------

extern "C" void kernel_launch(void* const* d_in, const int* in_sizes, int n_in,
                              void* d_out, int out_size, void* d_ws, size_t ws_size,
                              hipStream_t stream) {
    const float* x  = (const float*)d_in[0];   // [S,E]
    const float* Wq = (const float*)d_in[1];   // [E,V]
    const float* Wk = (const float*)d_in[2];
    const float* Wv = (const float*)d_in[3];
    float* out = (float*)d_out;                // [S,V]

    char* ws = (char*)d_ws;
    size_t off = 0;
    unsigned short* xb  = (unsigned short*)(ws + off); off += (size_t)S_LEN * E_DIM * 2;
    unsigned short* wqt = (unsigned short*)(ws + off); off += (size_t)E_DIM * V_DIM * 2; // [V,E]
    unsigned short* wkt = (unsigned short*)(ws + off); off += (size_t)E_DIM * V_DIM * 2;
    unsigned short* wvt = (unsigned short*)(ws + off); off += (size_t)E_DIM * V_DIM * 2;
    unsigned short* qb  = (unsigned short*)(ws + off); off += (size_t)S_LEN * V_DIM * 2; // [S,V]
    unsigned short* kb  = (unsigned short*)(ws + off); off += (size_t)S_LEN * V_DIM * 2; // [S,V]
    unsigned short* vt  = (unsigned short*)(ws + off); off += (size_t)S_LEN * V_DIM * 2; // [V,S]
    float* scores = (float*)(ws + off);        off += (size_t)S_LEN * S_LEN * 4;

    // ---- 1) casts (x plain; weights transposed to [V,E])
    {
        int n4 = (S_LEN * E_DIM) / 4;
        cast_f32_to_bf16<<<(n4 + 255) / 256, 256, 0, stream>>>((const float4*)x, (ushort4*)xb, n4);
        dim3 tg(V_DIM / 32, E_DIM / 32);
        cast_transpose_f32_bf16<<<tg, 256, 0, stream>>>(Wq, wqt, E_DIM, V_DIM);
        cast_transpose_f32_bf16<<<tg, 256, 0, stream>>>(Wk, wkt, E_DIM, V_DIM);
        cast_transpose_f32_bf16<<<tg, 256, 0, stream>>>(Wv, wvt, E_DIM, V_DIM);
    }

    dim3 blk(256);

    // ---- 2) projections: A=xb [S,E], B=W^T [V,E] (BT path)
    {
        dim3 grd(V_DIM / 128, S_LEN / 128);
        gemm_bt_bf16_wmma<1><<<grd, blk, 0, stream>>>(xb, E_DIM, wqt, E_DIM, qb, V_DIM, S_LEN, V_DIM, E_DIM);
        gemm_bt_bf16_wmma<1><<<grd, blk, 0, stream>>>(xb, E_DIM, wkt, E_DIM, kb, V_DIM, S_LEN, V_DIM, E_DIM);
        // V stored transposed: vt[V,S], ldc = S  (packed 16B stores)
        gemm_bt_bf16_wmma<2><<<grd, blk, 0, stream>>>(xb, E_DIM, wvt, E_DIM, vt, S_LEN, S_LEN, V_DIM, E_DIM);
    }

    // ---- 3) scores = Q @ K^T : A=qb [S,V], B=kb [S,V] (BT) -> f32 [S,S]
    {
        dim3 grd(S_LEN / 128, S_LEN / 128);
        gemm_bt_bf16_wmma<0><<<grd, blk, 0, stream>>>(qb, V_DIM, kb, V_DIM, scores, S_LEN, S_LEN, S_LEN, V_DIM);
    }

    // ---- 4) softmax rows, probs bf16 in-place (row stride 2*S bf16 elements)
    softmax_rows_inplace<<<S_LEN, 256, 0, stream>>>(scores, S_LEN);

    // ---- 5) z = P @ V : A=P bf16 (lda=2S), B=vt [V,S] (BT) -> f32 out
    {
        dim3 grd(V_DIM / 128, S_LEN / 128);
        gemm_bt_bf16_wmma<0><<<grd, blk, 0, stream>>>((const unsigned short*)scores, 2 * S_LEN,
                                                      vt, S_LEN, out, V_DIM,
                                                      S_LEN, V_DIM, S_LEN);
    }
}
